// MambaBlock_26869315404119
// MI455X (gfx1250) — compile-verified, matrix operand order checked by reading, not run on hardware
//
#include <hip/hip_runtime.h>
#include <hip/hip_bf16.h>

// ---------------- types for WMMA fragments ----------------
typedef __attribute__((ext_vector_type(16))) _Float16 v16h;
typedef __attribute__((ext_vector_type(8)))  _Float16 v8h;
typedef __attribute__((ext_vector_type(8)))  float    v8f;
typedef __attribute__((ext_vector_type(4)))  int      v4i;

union V16 { v16h v; v8h h[2]; };

#define B_SZ    2
#define L_SZ    1024
#define D_MODEL 1024
#define D_INNER 2048
#define D_STATE 16
#define DT_RANK 64
#define ROWS    (B_SZ * L_SZ)   // 2048

// CDNA5 async global->LDS path (guarded: fall back to load+ds_store staging)
#if defined(__has_builtin)
#  if __has_builtin(__builtin_amdgcn_global_load_async_to_lds_b128)
#    define HAVE_ASYNC_LDS 1
#  endif
#endif
#ifndef HAVE_ASYNC_LDS
#  define HAVE_ASYNC_LDS 0
#endif

typedef __attribute__((address_space(1))) v4i gv4i;   // global int4
typedef __attribute__((address_space(3))) v4i lv4i;   // LDS int4

__device__ __forceinline__ void wait_async_all() {
#if defined(__has_builtin) && __has_builtin(__builtin_amdgcn_s_wait_asynccnt)
    __builtin_amdgcn_s_wait_asynccnt(0);
#else
    asm volatile("s_wait_asynccnt 0x0" ::: "memory");
#endif
}

// copy 16 bytes global -> LDS (async on CDNA5, sync fallback otherwise)
__device__ __forceinline__ void copy16_g2l(const _Float16* gsrc, _Float16* ldst) {
#if HAVE_ASYNC_LDS
    __builtin_amdgcn_global_load_async_to_lds_b128((gv4i*)gsrc, (lv4i*)ldst, 0, 0);
#else
    *(v8h*)ldst = *(const v8h*)gsrc;
#endif
}

// ---------------- f32 -> f16 convert ----------------
__global__ void cvt_f32_f16(const float* __restrict__ in, _Float16* __restrict__ out, int n) {
    int i = blockIdx.x * blockDim.x + threadIdx.x;
    if (i < n) out[i] = (_Float16)in[i];
}

// ---------------- RMSNorm -> f16 activations ----------------
__global__ void rmsnorm_f16(const float* __restrict__ x, const float* __restrict__ w,
                            _Float16* __restrict__ xn) {
    const int D = D_MODEL;
    int row = blockIdx.x;
    const float* xr = x + (size_t)row * D;
    float s = 0.f;
    for (int i = threadIdx.x; i < D; i += blockDim.x) { float v = xr[i]; s += v * v; }
    __shared__ float red[32];
    for (int off = 16; off > 0; off >>= 1) s += __shfl_down(s, off, 32);
    int wid = threadIdx.x >> 5, lane = threadIdx.x & 31;
    if (lane == 0) red[wid] = s;
    __syncthreads();
    if (wid == 0) {
        float t = (lane < (int)(blockDim.x >> 5)) ? red[lane] : 0.f;
        for (int off = 16; off > 0; off >>= 1) t += __shfl_down(t, off, 32);
        if (lane == 0) red[0] = rsqrtf(t / (float)D + 1e-5f);
    }
    __syncthreads();
    float scale = red[0];
    _Float16* on = xn + (size_t)row * D;
    for (int i = threadIdx.x; i < D; i += blockDim.x)
        on[i] = (_Float16)(xr[i] * scale * w[i]);
}

// ---------------- single-tile NT GEMM via WMMA (for small N, e.g. x_proj) ----
__global__ void gemm_nt_wmma(const _Float16* __restrict__ A, const _Float16* __restrict__ Bw,
                             const float* __restrict__ residual, float* __restrict__ C,
                             int M, int N, int K) {
    int wavesPerBlock = blockDim.x >> 5;
    int waveId = threadIdx.x >> 5;
    int lane   = threadIdx.x & 31;
    int tilesN = N >> 4;
    int tilesM = M >> 4;
    int tileId = blockIdx.x * wavesPerBlock + waveId;
    if (tileId >= tilesM * tilesN) return;          // uniform per-wave
    int tm = tileId / tilesN;
    int tn = tileId % tilesN;

    int r    = lane & 15;
    int koff = (lane >> 4) << 3;

    const _Float16* arow = A  + (size_t)(tm * 16 + r) * K;
    const _Float16* brow = Bw + (size_t)(tn * 16 + r) * K;

    v8f acc = {};
    for (int k0 = 0; k0 < K; k0 += 32) {
        V16 a, b;
        a.h[0] = *(const v8h*)(arow + k0 + koff);
        a.h[1] = *(const v8h*)(arow + k0 + koff + 16);
        b.h[0] = *(const v8h*)(brow + k0 + koff);
        b.h[1] = *(const v8h*)(brow + k0 + koff + 16);
        acc = __builtin_amdgcn_wmma_f32_16x16x32_f16(false, a.v, false, b.v,
                                                     (short)0, acc, false, false);
    }
    int n     = lane & 15;
    int mbase = tm * 16 + ((lane >> 4) << 3);
    int nn    = tn * 16 + n;
#pragma unroll
    for (int rr = 0; rr < 8; ++rr) {
        int m = mbase + rr;
        float v = acc[rr];
        if (residual) v += residual[(size_t)m * N + nn];
        C[(size_t)m * N + nn] = v;
    }
}

// ---------------- blocked NT GEMM: 64x256 per block (4 waves x 64x64) -------
// A k-slab (64 rows x 32 halves, 4 KB) staged in LDS via async-to-LDS,
// double buffered; B fragments stream from global per wave.
#define BLK_M  64
#define BLK_N  256
#define SLAB_H (BLK_M * 32)    // halves per slab

__global__ void __launch_bounds__(128)
gemm_nt_wmma_blocked(const _Float16* __restrict__ A, const _Float16* __restrict__ Bw,
                     const float* __restrict__ residual, float* __restrict__ C,
                     int M, int N, int K) {
    __shared__ _Float16 As[2][SLAB_H];   // 2 x 4 KB

    int wave = threadIdx.x >> 5;
    int lane = threadIdx.x & 31;
    int mblk = blockIdx.y * BLK_M;
    int nblk = blockIdx.x * BLK_N + wave * 64;

    int r    = lane & 15;
    int koff = (lane >> 4) << 3;

    int t = threadIdx.x;    // 0..127; handles staging pieces t and t+128

    v8f acc[4][4];
#pragma unroll
    for (int i = 0; i < 4; ++i)
#pragma unroll
        for (int j = 0; j < 4; ++j) acc[i][j] = (v8f){};

    const int steps = K >> 5;   // K / 32

    // -------- prologue: stage slab 0 into buf 0 --------
#pragma unroll
    for (int pp = 0; pp < 2; ++pp) {
        int p   = t + pp * 128;
        int row = p >> 2, q = p & 3;
        copy16_g2l(A + (size_t)(mblk + row) * K + q * 8, &As[0][row * 32 + q * 8]);
    }

    for (int s = 0; s < steps; ++s) {
        int buf = s & 1;
#if HAVE_ASYNC_LDS
        wait_async_all();
#endif
        __syncthreads();

        // stage next slab into other buffer (overlaps with compute below)
        if (s + 1 < steps) {
            int k1 = (s + 1) << 5;
#pragma unroll
            for (int pp = 0; pp < 2; ++pp) {
                int p   = t + pp * 128;
                int row = p >> 2, q = p & 3;
                copy16_g2l(A + (size_t)(mblk + row) * K + k1 + q * 8,
                           &As[buf ^ 1][row * 32 + q * 8]);
            }
        }

        // A fragments for 4 M-subtiles from LDS (ds_load_b128 pairs)
        V16 af[4];
#pragma unroll
        for (int i = 0; i < 4; ++i) {
            const _Float16* lsrc = &As[buf][(i * 16 + r) * 32 + koff];
            af[i].h[0] = *(const v8h*)(lsrc);
            af[i].h[1] = *(const v8h*)(lsrc + 16);
        }

        int k0 = s << 5;
#pragma unroll
        for (int j = 0; j < 4; ++j) {
            V16 bf;
            const _Float16* brow = Bw + (size_t)(nblk + j * 16 + r) * K + k0 + koff;
            bf.h[0] = *(const v8h*)(brow);
            bf.h[1] = *(const v8h*)(brow + 16);
#pragma unroll
            for (int i = 0; i < 4; ++i) {
                acc[i][j] = __builtin_amdgcn_wmma_f32_16x16x32_f16(
                    false, af[i].v, false, bf.v, (short)0, acc[i][j], false, false);
            }
        }
    }

    // -------- store 64x64 per wave --------
    int n     = lane & 15;
    int mhalf = (lane >> 4) << 3;
#pragma unroll
    for (int i = 0; i < 4; ++i) {
#pragma unroll
        for (int j = 0; j < 4; ++j) {
            int col = nblk + j * 16 + n;
#pragma unroll
            for (int rr = 0; rr < 8; ++rr) {
                int m = mblk + i * 16 + mhalf + rr;
                float v = acc[i][j][rr];
                if (residual) v += residual[(size_t)m * N + col];
                C[(size_t)m * N + col] = v;
            }
        }
    }
}

// ---------------- depthwise conv (K=4, causal) + SiLU ----------------
__global__ void conv_silu(const float* __restrict__ xz, const float* __restrict__ cw,
                          const float* __restrict__ cb, float* __restrict__ u32,
                          _Float16* __restrict__ uh) {
    size_t idx = (size_t)blockIdx.x * blockDim.x + threadIdx.x;
    size_t total = (size_t)ROWS * D_INNER;
    if (idx >= total) return;
    int d = (int)(idx % D_INNER);
    size_t rl = idx / D_INNER;
    int l = (int)(rl % L_SZ);
    size_t brow = rl - l;
    float acc = cb[d];
#pragma unroll
    for (int j = 0; j < 4; ++j) {
        int li = l - 3 + j;
        if (li >= 0) acc += cw[d * 4 + j] * xz[(brow + (size_t)li) * (2 * D_INNER) + d];
    }
    float sv = acc / (1.f + __expf(-acc));
    u32[idx] = sv;
    uh[idx]  = (_Float16)sv;
}

// ---------------- dt = softplus(dt_in @ dt_proj_w^T + b) ----------------
__global__ void dt_softplus(const float* __restrict__ x_dbl, const float* __restrict__ dtw,
                            const float* __restrict__ dtb, float* __restrict__ dt) {
    size_t idx = (size_t)blockIdx.x * blockDim.x + threadIdx.x;
    size_t total = (size_t)ROWS * D_INNER;
    if (idx >= total) return;
    int d = (int)(idx % D_INNER);
    size_t rl = idx / D_INNER;
    const float* row = x_dbl + rl * 96;
    const float* wr  = dtw + (size_t)d * DT_RANK;
    float acc = dtb[d];
#pragma unroll 8
    for (int j = 0; j < DT_RANK; ++j) acc += row[j] * wr[j];
    dt[idx] = (acc > 20.f) ? acc : log1pf(__expf(acc));
}

// ---------------- selective scan over L (serial), 4096 channels parallel ----
__global__ void scan_kernel(const float* __restrict__ dt, const float* __restrict__ u32,
                            const float* __restrict__ x_dbl, const float* __restrict__ xz,
                            const float* __restrict__ A_log, const float* __restrict__ Dp,
                            _Float16* __restrict__ yh) {
    int gid = blockIdx.x * blockDim.x + threadIdx.x;
    if (gid >= B_SZ * D_INNER) return;
    int b = gid >> 11;
    int d = gid & (D_INNER - 1);

    float a[D_STATE], h[D_STATE];
#pragma unroll
    for (int n = 0; n < D_STATE; ++n) {
        a[n] = -__expf(A_log[(size_t)d * D_STATE + n]);
        h[n] = 0.f;
    }
    float Dd = Dp[d];

    for (int l = 0; l < L_SZ; ++l) {
        size_t rl = (size_t)b * L_SZ + l;
        if (l + 1 < L_SZ) {
            __builtin_prefetch(dt  + (rl + 1) * D_INNER + d, 0, 1);
            __builtin_prefetch(u32 + (rl + 1) * D_INNER + d, 0, 1);
        }
        float dtv = dt[rl * D_INNER + d];
        float uv  = u32[rl * D_INNER + d];
        const float* Brow = x_dbl + rl * 96 + DT_RANK;
        const float* Crow = Brow + D_STATE;
        float du = dtv * uv;
        float y = 0.f;
#pragma unroll
        for (int n = 0; n < D_STATE; ++n) {
            h[n] = __expf(dtv * a[n]) * h[n] + du * Brow[n];
            y += h[n] * Crow[n];
        }
        float z  = xz[rl * (2 * D_INNER) + D_INNER + d];
        float sz = z / (1.f + __expf(-z));
        yh[rl * D_INNER + d] = (_Float16)((y + uv * Dd) * sz);
    }
}

// ---------------- host-side launcher ----------------
extern "C" void kernel_launch(void* const* d_in, const int* in_sizes, int n_in,
                              void* d_out, int out_size, void* d_ws, size_t ws_size,
                              hipStream_t stream) {
    const float* x          = (const float*)d_in[0];
    const float* norm_w     = (const float*)d_in[1];
    const float* in_proj_w  = (const float*)d_in[2];
    const float* conv_w     = (const float*)d_in[3];
    const float* conv_b     = (const float*)d_in[4];
    const float* x_proj_w   = (const float*)d_in[5];
    const float* dt_proj_w  = (const float*)d_in[6];
    const float* dt_proj_b  = (const float*)d_in[7];
    const float* A_log      = (const float*)d_in[8];
    const float* Dp         = (const float*)d_in[9];
    const float* out_proj_w = (const float*)d_in[10];
    float* out = (float*)d_out;

    char* ws = (char*)d_ws;
    size_t off = 0;
    _Float16* xn_h   = (_Float16*)(ws + off); off += (size_t)ROWS * D_MODEL * 2;
    _Float16* win_h  = (_Float16*)(ws + off); off += (size_t)(2*D_INNER) * D_MODEL * 2;
    float*    xz     = (float*)   (ws + off); off += (size_t)ROWS * (2*D_INNER) * 4;
    float*    u32    = (float*)   (ws + off); off += (size_t)ROWS * D_INNER * 4;
    _Float16* uh     = (_Float16*)(ws + off); off += (size_t)ROWS * D_INNER * 2;
    _Float16* wxp_h  = (_Float16*)(ws + off); off += (size_t)96 * D_INNER * 2;
    float*    x_dbl  = (float*)   (ws + off); off += (size_t)ROWS * 96 * 4;
    float*    dtbuf  = (float*)   (ws + off); off += (size_t)ROWS * D_INNER * 4;
    _Float16* yh     = (_Float16*)(ws + off); off += (size_t)ROWS * D_INNER * 2;
    _Float16* wout_h = (_Float16*)(ws + off); off += (size_t)D_MODEL * D_INNER * 2;

    // 1. weight conversions to f16
    {
        int n1 = 2 * D_INNER * D_MODEL;
        cvt_f32_f16<<<(n1 + 255) / 256, 256, 0, stream>>>(in_proj_w, win_h, n1);
        int n2 = 96 * D_INNER;
        cvt_f32_f16<<<(n2 + 255) / 256, 256, 0, stream>>>(x_proj_w, wxp_h, n2);
        int n3 = D_MODEL * D_INNER;
        cvt_f32_f16<<<(n3 + 255) / 256, 256, 0, stream>>>(out_proj_w, wout_h, n3);
    }

    // 2. RMSNorm -> f16
    rmsnorm_f16<<<ROWS, 256, 0, stream>>>(x, norm_w, xn_h);

    // 3. in_proj: (2048 x 1024) x (4096 x 1024)^T -> xz (2048 x 4096)
    {
        dim3 grid((2 * D_INNER) / BLK_N, ROWS / BLK_M);   // 16 x 32
        gemm_nt_wmma_blocked<<<grid, 128, 0, stream>>>(xn_h, win_h, nullptr, xz,
                                                       ROWS, 2 * D_INNER, D_MODEL);
    }

    // 4. causal depthwise conv + SiLU -> u (f32 + f16)
    {
        size_t total = (size_t)ROWS * D_INNER;
        conv_silu<<<(int)((total + 255) / 256), 256, 0, stream>>>(xz, conv_w, conv_b, u32, uh);
    }

    // 5. x_proj: (2048 x 2048) x (96 x 2048)^T -> x_dbl (2048 x 96)
    {
        int tiles = (ROWS / 16) * (96 / 16);
        gemm_nt_wmma<<<(tiles + 3) / 4, 128, 0, stream>>>(uh, wxp_h, nullptr, x_dbl,
                                                          ROWS, 96, D_INNER);
    }

    // 6. dt projection + softplus
    {
        size_t total = (size_t)ROWS * D_INNER;
        dt_softplus<<<(int)((total + 255) / 256), 256, 0, stream>>>(x_dbl, dt_proj_w, dt_proj_b, dtbuf);
    }

    // 7. selective scan
    scan_kernel<<<(B_SZ * D_INNER) / 256, 256, 0, stream>>>(dtbuf, u32, x_dbl, xz, A_log, Dp, yh);

    // 8. out_proj + residual
    {
        dim3 grid(D_MODEL / BLK_N, ROWS / BLK_M);          // 4 x 32
        gemm_nt_wmma_blocked<<<grid, 128, 0, stream>>>(yh, wout_h, x, out,
                                                       ROWS, D_MODEL, D_INNER);
    }

    (void)in_sizes; (void)n_in; (void)out_size; (void)ws_size; (void)off;
}
